// CausalSelfAttention_82231443849665
// MI455X (gfx1250) — matrix operand order repaired and checked by reading, not compiled, not run on hardware
//
#include <hip/hip_runtime.h>
#include <hip/hip_bf16.h>
#include <stdint.h>

#define B_SZ    4
#define T_SEQ   2048
#define D_MODEL 1024
#define N_HEAD  16
#define HD      64
#define M_ROWS  (B_SZ * T_SEQ)   // 8192

typedef __attribute__((ext_vector_type(16))) __bf16 v16bf;
typedef __attribute__((ext_vector_type(8)))  float  v8f;
typedef unsigned short bf16_u;   // bf16 bits as storage

union FragU { v16bf v; uint4 u[2]; };

__device__ __forceinline__ bf16_u f2bf(float f) {
  unsigned u = __float_as_uint(f);
  u += 0x7FFFu + ((u >> 16) & 1u);          // round-to-nearest-even
  return (bf16_u)(u >> 16);
}
__device__ __forceinline__ float bf2f(bf16_u h) {
  return __uint_as_float(((unsigned)h) << 16);
}

// A fragment (16x32 bf16, M x K): lane<16 -> row=lane, k in {0..7, 16..23};
// lane>=16 -> row=lane-16, k in {8..15, 24..31}.  base = &A[row0][k0], ld in elems.
__device__ __forceinline__ v16bf load_frag_a(const bf16_u* base, int ld) {
  int lane = threadIdx.x & 31;
  const bf16_u* p = base + (size_t)(lane & 15) * ld + ((lane >> 4) << 3);
  FragU r;
  r.u[0] = *(const uint4*)(p);        // k offsets +0..7
  r.u[1] = *(const uint4*)(p + 16);   // k offsets +16..23
  return r.v;
}

// B fragment (32x16 bf16, K x N), stored n-major (row n, contiguous k):
// lane<16 -> n=lane, k in {0..7, 16..23} ; lane>=16 -> n=lane-16, k in {8..15, 24..31}.
__device__ __forceinline__ v16bf load_frag_b(const bf16_u* base, int ld) {
  int lane = threadIdx.x & 31;
  const bf16_u* p = base + (size_t)(lane & 15) * ld + ((lane >> 4) << 3);
  FragU r;
  r.u[0] = *(const uint4*)(p);
  r.u[1] = *(const uint4*)(p + 16);
  return r.v;
}

__device__ __forceinline__ v8f wmma_bf16(v16bf a, v16bf b, v8f c) {
  return __builtin_amdgcn_wmma_f32_16x16x32_bf16(false, a, false, b, (short)0, c,
                                                 false, false);
}

// CDNA5 async copy: global -> LDS, no VGPR staging, tracked by ASYNCcnt.
__device__ __forceinline__ void async_load_b128(unsigned lds_addr, const void* gaddr) {
  asm volatile("global_load_async_to_lds_b128 %0, %1, off"
               :: "v"(lds_addr), "v"((unsigned long long)(uintptr_t)gaddr)
               : "memory");
}
__device__ __forceinline__ void wait_async0() {
  asm volatile("s_wait_asynccnt 0" ::: "memory");
}

// ---------------------------------------------------------------- convert
__global__ void cvt_kernel(const float* __restrict__ in, bf16_u* __restrict__ out, int n) {
  int i = blockIdx.x * blockDim.x + threadIdx.x;
  if (i < n) out[i] = f2bf(in[i]);
}

// ---------------------------------------------------------------- GEMM  Y = A (MxK) * W^T (W is NxK row-major)
// Block = 4 waves sharing one 32-col B tile (async double-buffered in LDS);
// each wave computes a 64x32 slice -> 256x32 per block.
// mode 0: bf16 out to [B][H][T][Hd];  mode 1: bf16 out to [B][H][Hd][T];
// mode 2: f32 row-major.
__global__ __launch_bounds__(128) void gemm_bf16_kernel(
    const bf16_u* __restrict__ A, const bf16_u* __restrict__ W,
    void* __restrict__ out, int M, int N, int K, int mode)
{
  __shared__ __attribute__((aligned(16))) bf16_u sB[2][32][32];  // [buf][n][k]

  int lane = threadIdx.x & 31;
  int wv   = threadIdx.x >> 5;
  int tn = N >> 5;
  int bm = blockIdx.x / tn, bn = blockIdx.x - bm * tn;
  int m0 = bm * 256 + wv * 64;
  int n0 = bn << 5;

  // per-thread async B-tile slice: 16 bytes = 8 bf16
  int tid = threadIdx.x;
  int brow = tid >> 2;             // n within tile (0..31)
  int bk   = (tid & 3) << 3;       // k offset within tile (0,8,16,24)
  const bf16_u* wsrc = W + (size_t)(n0 + brow) * K + bk;

  async_load_b128((unsigned)(uintptr_t)&sB[0][brow][bk], wsrc);   // k0 = 0

  v8f c[4][2] = {};
  const bf16_u* Ab = A + (size_t)m0 * K;

  for (int k0 = 0; k0 < K; k0 += 32) {
    int buf = (k0 >> 5) & 1;
    wait_async0();                 // own wave's async issue complete (LDS written)
    __syncthreads();               // everyone's tile writes visible; prev readers done
    if (k0 + 32 < K)
      async_load_b128((unsigned)(uintptr_t)&sB[buf ^ 1][brow][bk], wsrc + k0 + 32);

    if (k0 + 64 < K)
      __builtin_prefetch((const void*)(Ab + k0 + 64), 0, 1);

    v16bf a0 = load_frag_a(Ab + (size_t)0  * K + k0, K);
    v16bf a1 = load_frag_a(Ab + (size_t)16 * K + k0, K);
    v16bf a2 = load_frag_a(Ab + (size_t)32 * K + k0, K);
    v16bf a3 = load_frag_a(Ab + (size_t)48 * K + k0, K);
    v16bf b0 = load_frag_b(&sB[buf][0][0],  32);   // n = 0..15
    v16bf b1 = load_frag_b(&sB[buf][16][0], 32);   // n = 16..31
    c[0][0] = wmma_bf16(a0, b0, c[0][0]);
    c[0][1] = wmma_bf16(a0, b1, c[0][1]);
    c[1][0] = wmma_bf16(a1, b0, c[1][0]);
    c[1][1] = wmma_bf16(a1, b1, c[1][1]);
    c[2][0] = wmma_bf16(a2, b0, c[2][0]);
    c[2][1] = wmma_bf16(a2, b1, c[2][1]);
    c[3][0] = wmma_bf16(a3, b0, c[3][0]);
    c[3][1] = wmma_bf16(a3, b1, c[3][1]);
  }

  // C layout: VGPR r -> (m = r or r+8 by lane half, n = lane&15)
  int nlo = lane & 15;
  int mb  = (lane >> 4) << 3;
  for (int i = 0; i < 4; ++i)
    for (int j = 0; j < 2; ++j)
      for (int r = 0; r < 8; ++r) {
        int m  = m0 + i * 16 + mb + r;
        int nn = n0 + j * 16 + nlo;
        float val = c[i][j][r];
        if (mode == 2) {
          ((float*)out)[(size_t)m * N + nn] = val;
        } else {
          int b = m / T_SEQ, t = m - b * T_SEQ;
          int h = nn >> 6, d = nn & 63;
          size_t idx;
          if (mode == 0) idx = (((size_t)(b * N_HEAD + h) * T_SEQ) + t) * HD + d;
          else           idx = (((size_t)(b * N_HEAD + h) * HD) + d) * T_SEQ + t;
          ((bf16_u*)out)[idx] = f2bf(val);
        }
      }
}

// ---------------------------------------------------------------- RoPE in place on [B][H][T][Hd]
__global__ void rope_kernel(bf16_u* __restrict__ q, bf16_u* __restrict__ k,
                            const float* __restrict__ cosb, const float* __restrict__ sinb)
{
  int idx = blockIdx.x * blockDim.x + threadIdx.x;   // (bh, t, d2<32)
  if (idx >= B_SZ * N_HEAD * T_SEQ * 32) return;
  int d2 = idx & 31;
  int t  = (idx >> 5) & (T_SEQ - 1);
  int bh = idx >> 16;                                // 32*2048 = 2^16
  size_t base = ((size_t)bh * T_SEQ + t) * HD;
  float c = cosb[t * HD + d2];
  float s = sinb[t * HD + d2];
  float q1 = bf2f(q[base + d2]), q2 = bf2f(q[base + d2 + 32]);
  q[base + d2]      = f2bf(q1 * c - q2 * s);
  q[base + d2 + 32] = f2bf(q2 * c + q1 * s);
  float k1 = bf2f(k[base + d2]), k2 = bf2f(k[base + d2 + 32]);
  k[base + d2]      = f2bf(k1 * c - k2 * s);
  k[base + d2 + 32] = f2bf(k2 * c + k1 * s);
}

// ---------------------------------------------------------------- flash attention
// one wave = one (b,h, 16-row q block). LDS traffic is wave-private; CDNA5
// split counters: wait ds/load/store before cross-lane reuse.
#define WAVE_SYNC() \
  asm volatile("s_wait_dscnt 0\n\ts_wait_storecnt 0\n\ts_wait_loadcnt 0" ::: "memory")

__global__ __launch_bounds__(128) void attn_kernel(
    const bf16_u* __restrict__ q, const bf16_u* __restrict__ k,
    const bf16_u* __restrict__ vt, bf16_u* __restrict__ o_attn)
{
  __shared__ float  sS[4][16][33];
  __shared__ __attribute__((aligned(16))) bf16_u sP[4][16][32];
  __shared__ float  sScale[4][16];
  __shared__ float  sL[4][16];

  int lane = threadIdx.x & 31;
  int wv   = threadIdx.x >> 5;
  int gw   = blockIdx.x * 4 + wv;
  int qb   = gw & (T_SEQ / 16 - 1);
  int bh   = gw >> 7;
  if (bh >= B_SZ * N_HEAD) return;
  int q0 = qb << 4;

  const bf16_u* qbase = q  + ((size_t)bh * T_SEQ + q0) * HD;
  const bf16_u* kbase = k  + (size_t)bh * T_SEQ * HD;
  const bf16_u* vbase = vt + (size_t)bh * HD * T_SEQ;

  v16bf qa0 = load_frag_a(qbase, HD);        // d = 0..31
  v16bf qa1 = load_frag_a(qbase + 32, HD);   // d = 32..63

  v8f o0 = {}, o1 = {}, o2 = {}, o3 = {};
  float m_i = -__builtin_inff();
  float l_i = 0.0f;

  int nlo = lane & 15;
  int mb  = (lane >> 4) << 3;

  int nkb = (q0 + 16 + 31) >> 5;             // causal: stop at diagonal
  for (int kb = 0; kb < nkb; ++kb) {
    int k0 = kb << 5;
    // S = Q * K^T, two 16x16 tiles over 32 k-columns
    v8f s0 = {}, s1 = {};
    {
      v16bf b00 = load_frag_b(kbase + (size_t)(k0)      * HD,      HD);
      v16bf b01 = load_frag_b(kbase + (size_t)(k0)      * HD + 32, HD);
      v16bf b10 = load_frag_b(kbase + (size_t)(k0 + 16) * HD,      HD);
      v16bf b11 = load_frag_b(kbase + (size_t)(k0 + 16) * HD + 32, HD);
      s0 = wmma_bf16(qa0, b00, s0);
      s0 = wmma_bf16(qa1, b01, s0);
      s1 = wmma_bf16(qa0, b10, s1);
      s1 = wmma_bf16(qa1, b11, s1);
    }
    for (int r = 0; r < 8; ++r) {
      sS[wv][mb + r][nlo]      = s0[r];
      sS[wv][mb + r][16 + nlo] = s1[r];
    }
    WAVE_SYNC();

    if (lane < 16) {                         // one lane per q-row: online softmax
      int row = lane, qpos = q0 + row;
      float mmax = m_i;
      float sv[32];
      for (int j = 0; j < 32; ++j) {
        float x = sS[wv][row][j] * 0.125f;   // Hd^-0.5
        sv[j] = (k0 + j <= qpos) ? x : -__builtin_inff();
        mmax = fmaxf(mmax, sv[j]);
      }
      float scl = __expf(m_i - mmax);
      float lsum = 0.0f;
      for (int j = 0; j < 32; ++j) {
        float p = __expf(sv[j] - mmax);      // masked -> exp(-inf)=0
        lsum += p;
        sP[wv][row][j] = f2bf(p);
      }
      l_i = l_i * scl + lsum;
      m_i = mmax;
      sScale[wv][row] = scl;
    }
    WAVE_SYNC();

    for (int r = 0; r < 8; ++r) {            // rescale running accumulators
      float f = sScale[wv][mb + r];
      o0[r] *= f; o1[r] *= f; o2[r] *= f; o3[r] *= f;
    }
    v16bf pa  = load_frag_a(&sP[wv][0][0], 32);            // P as A-fragment
    v16bf bv0 = load_frag_b(vbase + k0, T_SEQ);            // V^T: rows=d, contiguous t
    v16bf bv1 = load_frag_b(vbase + (size_t)16 * T_SEQ + k0, T_SEQ);
    v16bf bv2 = load_frag_b(vbase + (size_t)32 * T_SEQ + k0, T_SEQ);
    v16bf bv3 = load_frag_b(vbase + (size_t)48 * T_SEQ + k0, T_SEQ);
    o0 = wmma_bf16(pa, bv0, o0);
    o1 = wmma_bf16(pa, bv1, o1);
    o2 = wmma_bf16(pa, bv2, o2);
    o3 = wmma_bf16(pa, bv3, o3);
  }

  if (lane < 16) sL[wv][lane] = l_i;
  WAVE_SYNC();

  int b = bh >> 4, h = bh & 15;
  for (int r = 0; r < 8; ++r) {
    float inv = 1.0f / sL[wv][mb + r];
    size_t rowbase = (((size_t)(b * T_SEQ + q0 + mb + r)) * N_HEAD + h) * HD;
    o_attn[rowbase +  0 + nlo] = f2bf(o0[r] * inv);
    o_attn[rowbase + 16 + nlo] = f2bf(o1[r] * inv);
    o_attn[rowbase + 32 + nlo] = f2bf(o2[r] * inv);
    o_attn[rowbase + 48 + nlo] = f2bf(o3[r] * inv);
  }
}

// ---------------------------------------------------------------- launch
extern "C" void kernel_launch(void* const* d_in, const int* in_sizes, int n_in,
                              void* d_out, int out_size, void* d_ws, size_t ws_size,
                              hipStream_t stream)
{
  const float* x    = (const float*)d_in[0];
  const float* cosb = (const float*)d_in[1];
  const float* sinb = (const float*)d_in[2];
  const float* wq   = (const float*)d_in[3];
  const float* wk   = (const float*)d_in[4];
  const float* wv   = (const float*)d_in[5];
  const float* wo   = (const float*)d_in[6];

  char* ws = (char*)d_ws;
  size_t off = 0;
  auto alloc = [&](size_t bytes) -> void* {
    void* p = (void*)(ws + off);
    off += (bytes + 255) & ~(size_t)255;
    return p;
  };
  const size_t M = M_ROWS, D = D_MODEL;
  bf16_u* xb  = (bf16_u*)alloc(M * D * 2);
  bf16_u* wqb = (bf16_u*)alloc(D * D * 2);
  bf16_u* wkb = (bf16_u*)alloc(D * D * 2);
  bf16_u* wvb = (bf16_u*)alloc(D * D * 2);
  bf16_u* wob = (bf16_u*)alloc(D * D * 2);
  bf16_u* qh  = (bf16_u*)alloc(M * D * 2);   // [B][H][T][Hd]
  bf16_u* kh  = (bf16_u*)alloc(M * D * 2);   // [B][H][T][Hd]
  bf16_u* vth = (bf16_u*)alloc(M * D * 2);   // [B][H][Hd][T]
  bf16_u* ah  = (bf16_u*)alloc(M * D * 2);   // [B][T][H][Hd] == row-major [M][D]

  int nx = (int)(M * D);
  int nw = (int)(D * D);
  cvt_kernel<<<(nx + 255) / 256, 256, 0, stream>>>(x,  xb,  nx);
  cvt_kernel<<<(nw + 255) / 256, 256, 0, stream>>>(wq, wqb, nw);
  cvt_kernel<<<(nw + 255) / 256, 256, 0, stream>>>(wk, wkb, nw);
  cvt_kernel<<<(nw + 255) / 256, 256, 0, stream>>>(wv, wvb, nw);
  cvt_kernel<<<(nw + 255) / 256, 256, 0, stream>>>(wo, wob, nw);

  int gemm_blocks = (int)((M / 256) * (D / 32));   // 32*32 = 1024
  gemm_bf16_kernel<<<gemm_blocks, 128, 0, stream>>>(xb, wqb, qh,  (int)M, (int)D, (int)D, 0);
  gemm_bf16_kernel<<<gemm_blocks, 128, 0, stream>>>(xb, wkb, kh,  (int)M, (int)D, (int)D, 0);
  gemm_bf16_kernel<<<gemm_blocks, 128, 0, stream>>>(xb, wvb, vth, (int)M, (int)D, (int)D, 1);

  int nr = B_SZ * N_HEAD * T_SEQ * 32;
  rope_kernel<<<(nr + 255) / 256, 256, 0, stream>>>(qh, kh, cosb, sinb);

  int attn_waves = B_SZ * N_HEAD * (T_SEQ / 16);   // 8192
  attn_kernel<<<attn_waves / 4, 128, 0, stream>>>(qh, kh, vth, ah);

  gemm_bf16_kernel<<<gemm_blocks, 128, 0, stream>>>(ah, wob, d_out, (int)M, (int)D, (int)D, 2);
}
